// Transformer_End_22488448762263
// MI455X (gfx1250) — matrix operand order repaired and checked by reading, not compiled, not run on hardware
//
#include <hip/hip_runtime.h>
#include <hip/hip_bf16.h>

// ---------------------------------------------------------------------------
// Nystrom-attention transformer for MI455X (gfx1250, wave32, WMMA).
// All GEMMs go through v_wmma_f32_16x16x32_f16 (f16 in, f32 accumulate).
// Staging: global_load_b128 -> v_cvt -> ds_store_b64, batched so loads issue
// back-to-back before the first wait.
// ---------------------------------------------------------------------------

typedef __attribute__((ext_vector_type(16))) _Float16 v16h;
typedef __attribute__((ext_vector_type(8)))  _Float16 v8h;
typedef __attribute__((ext_vector_type(4)))  _Float16 v4h;
typedef __attribute__((ext_vector_type(8)))  float    v8f;

union V16H { v16h v; v8h h[2]; };

#define BDIM  4        // batch
#define SEQN  4096     // raw sequence
#define NP    4097     // +cls token
#define PADL  255      // left pad to multiple of 256
#define NPD   4352     // padded sequence (= 34*128)
#define DIM   512
#define HEADS 8
#define DH    64
#define MLM   256      // landmarks
#define LGRP  17       // NPD / MLM
#define KSZ   33       // conv kernel
#define QKVD  1536
#define OUTER 1024

#define TM 128
#define TN 64
#define TK 32
#define LDSTR 40       // halves per LDS row (80 B; 8B/16B-aligned chunks)

// ---------------------------------------------------------------------------
// Batched WMMA GEMM:  C = alpha * A @ op(B)  (+ epilogue)
//   TRB=1: op(B)=B^T with B stored [N,K];  TRB=0: op(B)=B stored [K,N]
//   EPI=0: store;  EPI=1: store relu(x+bias);  EPI=2: C[row-skip] += x + bias
// Per-batch pointers: base + b*s?b + h*s?h with b=z/HB, h=z%HB.
// Requirements (hold for every GEMM here): M%128==0, N%64==0, K%32==0,
// all leading dims / base offsets multiples of 4 floats (16B alignment).
// ---------------------------------------------------------------------------
struct GemmP {
  const float* A; const float* B; float* C; const float* bias;
  int M, N, K, lda, ldb, ldc, HB, skip;
  long long sAb, sAh, sBb, sBh, sCb, sCh;
  float alpha;
};

template<int TRB, int EPI>
__global__ __launch_bounds__(256) void gemm_wmma_k(GemmP p) {
  __shared__ __align__(16) _Float16 lA[TM * LDSTR];
  __shared__ __align__(16) _Float16 lB[TN * LDSTR];
  const int z  = blockIdx.z;
  const int bb = z / p.HB, hh = z % p.HB;
  const float* A = p.A + (long long)bb * p.sAb + (long long)hh * p.sAh;
  const float* B = p.B + (long long)bb * p.sBb + (long long)hh * p.sBh;
  float*       C = p.C + (long long)bb * p.sCb + (long long)hh * p.sCh;
  const int tMo = blockIdx.y * TM, tNo = blockIdx.x * TN;
  const int tid = threadIdx.x, lane = tid & 31, wave = tid >> 5;
  const int wm = (wave >> 1) * 32, wn = (wave & 1) * 32;   // 8 waves: 4x2
  const int mrow = lane & 15;
  const int aoff = (lane >> 4) * 8;    // A frag: K {0..7,16..23} / {8..15,24..31}
  const int boff = (lane >> 4) * 8;    // B frag: K {0..7,16..23} / {8..15,24..31}
  v8f acc[2][2] = {};

  for (int kt = 0; kt < p.K; kt += TK) {
    __syncthreads();
    // ---- stage A tile 128x32 (1024 float4, 4 per thread) ----
    float4 va[4];
#pragma unroll
    for (int i = 0; i < 4; ++i) {
      int q = tid + i * 256, r = q >> 3, c4 = q & 7;
      va[i] = *(const float4*)&A[(long long)(tMo + r) * p.lda + kt + c4 * 4];
    }
    // ---- stage B tile (512 float4, 2 per thread) ----
    float4 vb[2];
    if (TRB) {
#pragma unroll
      for (int i = 0; i < 2; ++i) {
        int q = tid + i * 256, r = q >> 3, c4 = q & 7;
        vb[i] = *(const float4*)&B[(long long)(tNo + r) * p.ldb + kt + c4 * 4];
      }
    } else {
#pragma unroll
      for (int i = 0; i < 2; ++i) {
        int q = tid + i * 256, kk = q >> 4, n4 = q & 15;
        vb[i] = *(const float4*)&B[(long long)(kt + kk) * p.ldb + tNo + n4 * 4];
      }
    }
    // ---- convert + LDS store ----
#pragma unroll
    for (int i = 0; i < 4; ++i) {
      int q = tid + i * 256, r = q >> 3, c4 = q & 7;
      v4h pk;
      pk[0] = (_Float16)va[i].x; pk[1] = (_Float16)va[i].y;
      pk[2] = (_Float16)va[i].z; pk[3] = (_Float16)va[i].w;
      *(v4h*)&lA[r * LDSTR + c4 * 4] = pk;        // ds_store_b64
    }
    if (TRB) {
#pragma unroll
      for (int i = 0; i < 2; ++i) {
        int q = tid + i * 256, r = q >> 3, c4 = q & 7;
        v4h pk;
        pk[0] = (_Float16)vb[i].x; pk[1] = (_Float16)vb[i].y;
        pk[2] = (_Float16)vb[i].z; pk[3] = (_Float16)vb[i].w;
        *(v4h*)&lB[r * LDSTR + c4 * 4] = pk;      // ds_store_b64
      }
    } else {
#pragma unroll
      for (int i = 0; i < 2; ++i) {               // transpose into [n][k]
        int q = tid + i * 256, kk = q >> 4, n4 = q & 15;
        lB[(n4 * 4 + 0) * LDSTR + kk] = (_Float16)vb[i].x;
        lB[(n4 * 4 + 1) * LDSTR + kk] = (_Float16)vb[i].y;
        lB[(n4 * 4 + 2) * LDSTR + kk] = (_Float16)vb[i].z;
        lB[(n4 * 4 + 3) * LDSTR + kk] = (_Float16)vb[i].w;
      }
    }
    __syncthreads();
    if (kt + TK < p.K) {                          // gfx1250 global_prefetch_b8
      if (tid < 128)
        __builtin_prefetch(&A[(long long)(tMo + tid) * p.lda + kt + TK], 0, 1);
      else if (TRB) {
        if (tid < 192)
          __builtin_prefetch(&B[(long long)(tNo + tid - 128) * p.ldb + kt + TK], 0, 1);
      } else if (tid < 160)
        __builtin_prefetch(&B[(long long)(kt + TK + tid - 128) * p.ldb + tNo], 0, 1);
    }
    V16H af[2], bf[2];
#pragma unroll
    for (int i = 0; i < 2; ++i) {
      int r = wm + i * 16 + mrow;
      af[i].h[0] = *(const v8h*)&lA[r * LDSTR + aoff];
      af[i].h[1] = *(const v8h*)&lA[r * LDSTR + 16 + aoff];
      int n = wn + i * 16 + mrow;
      bf[i].h[0] = *(const v8h*)&lB[n * LDSTR + boff];
      bf[i].h[1] = *(const v8h*)&lB[n * LDSTR + boff + 16];
    }
#pragma unroll
    for (int i = 0; i < 2; ++i)
#pragma unroll
      for (int j = 0; j < 2; ++j)
        acc[i][j] = __builtin_amdgcn_wmma_f32_16x16x32_f16(
            false, af[i].v, false, bf[j].v, (short)0, acc[i][j], false, false);
  }

  // Epilogue. C/D layout: lane<16 -> N=lane, M=e; lane>=16 -> N=lane-16, M=8+e.
#pragma unroll
  for (int i = 0; i < 2; ++i) {
#pragma unroll
    for (int j = 0; j < 2; ++j) {
      int col = tNo + wn + j * 16 + (lane & 15);
      float bv = (EPI >= 1 && p.bias) ? p.bias[col] : 0.f;
#pragma unroll
      for (int e = 0; e < 8; ++e) {
        int row = tMo + wm + i * 16 + (lane >> 4) * 8 + e;
        float vv = acc[i][j][e] * p.alpha;
        if (EPI == 0) {
          C[(long long)row * p.ldc + col] = vv;
        } else if (EPI == 1) {
          vv += bv; if (vv < 0.f) vv = 0.f;
          C[(long long)row * p.ldc + col] = vv;
        } else {
          if (row >= p.skip)
            C[(long long)(row - p.skip) * p.ldc + col] += vv + bv;
        }
      }
    }
  }
}

template<int TRB, int EPI>
static void launch_gemm(hipStream_t st, const float* A, const float* B, float* C,
                        const float* bias, int M, int N, int K,
                        int lda, int ldb, int ldc,
                        long long sAb, long long sAh, long long sBb, long long sBh,
                        long long sCb, long long sCh, int HB, int batches,
                        float alpha, int skip) {
  GemmP p{A, B, C, bias, M, N, K, lda, ldb, ldc, HB, skip,
          sAb, sAh, sBb, sBh, sCb, sCh, alpha};
  gemm_wmma_k<TRB, EPI><<<dim3(N / TN, M / TM, batches), 256, 0, st>>>(p);
}

// ---------------------------------------------------------------------------
// LayerNorm into left-zero-padded buffer: xpad[b, 0..PADL) = 0,
// xpad[b, PADL+r] = LN(h[b, r]) * g + b.   (DIM=512, block=256)
// ---------------------------------------------------------------------------
__global__ __launch_bounds__(256) void ln_pad_k(const float* __restrict__ h,
                                                const float* __restrict__ g,
                                                const float* __restrict__ be,
                                                float* __restrict__ xpad) {
  const int row = blockIdx.x, bb = blockIdx.y, tid = threadIdx.x;
  float* o = xpad + ((long long)bb * NPD + row) * DIM;
  if (row < PADL) { o[tid] = 0.f; o[tid + 256] = 0.f; return; }
  const float* x = h + ((long long)bb * NP + (row - PADL)) * DIM;
  float v0 = x[tid], v1 = x[tid + 256];
  float s = v0 + v1, s2 = v0 * v0 + v1 * v1;
  for (int o2 = 16; o2; o2 >>= 1) {
    s  += __shfl_down(s, o2, 32);
    s2 += __shfl_down(s2, o2, 32);
  }
  __shared__ float rs[8], rq[8];
  if ((tid & 31) == 0) { rs[tid >> 5] = s; rq[tid >> 5] = s2; }
  __syncthreads();
  float ts = rs[0]+rs[1]+rs[2]+rs[3]+rs[4]+rs[5]+rs[6]+rs[7];
  float tq = rq[0]+rq[1]+rq[2]+rq[3]+rq[4]+rq[5]+rq[6]+rq[7];
  float mu  = ts * (1.f / DIM);
  float inv = rsqrtf(tq * (1.f / DIM) - mu * mu + 1e-5f);
  o[tid]       = (v0 - mu) * inv * g[tid]       + be[tid];
  o[tid + 256] = (v1 - mu) * inv * g[tid + 256] + be[tid + 256];
}

// Final LN of the cls row only -> out[b*DIM..]
__global__ __launch_bounds__(256) void ln_row0_k(const float* __restrict__ h,
                                                 const float* __restrict__ g,
                                                 const float* __restrict__ be,
                                                 float* __restrict__ out) {
  const int bb = blockIdx.x, tid = threadIdx.x;
  const float* x = h + (long long)bb * NP * DIM;
  float v0 = x[tid], v1 = x[tid + 256];
  float s = v0 + v1, s2 = v0 * v0 + v1 * v1;
  for (int o2 = 16; o2; o2 >>= 1) {
    s  += __shfl_down(s, o2, 32);
    s2 += __shfl_down(s2, o2, 32);
  }
  __shared__ float rs[8], rq[8];
  if ((tid & 31) == 0) { rs[tid >> 5] = s; rq[tid >> 5] = s2; }
  __syncthreads();
  float ts = rs[0]+rs[1]+rs[2]+rs[3]+rs[4]+rs[5]+rs[6]+rs[7];
  float tq = rq[0]+rq[1]+rq[2]+rq[3]+rq[4]+rq[5]+rq[6]+rq[7];
  float mu  = ts * (1.f / DIM);
  float inv = rsqrtf(tq * (1.f / DIM) - mu * mu + 1e-5f);
  float* o = out + bb * DIM;
  o[tid]       = (v0 - mu) * inv * g[tid]       + be[tid];
  o[tid + 256] = (v1 - mu) * inv * g[tid + 256] + be[tid + 256];
}

// Row-wise in-place softmax, one block per row of length L.
__global__ __launch_bounds__(256) void softmax_k(float* __restrict__ x, int L) {
  float* pr = x + (long long)blockIdx.x * L;
  const int tid = threadIdx.x;
  __shared__ float rm[8], rsum[8];
  float mx = -3.4e38f;
  for (int c = tid; c < L; c += 256) mx = fmaxf(mx, pr[c]);
  for (int o = 16; o; o >>= 1) mx = fmaxf(mx, __shfl_down(mx, o, 32));
  if ((tid & 31) == 0) rm[tid >> 5] = mx;
  __syncthreads();
  float m = fmaxf(fmaxf(fmaxf(rm[0], rm[1]), fmaxf(rm[2], rm[3])),
                  fmaxf(fmaxf(rm[4], rm[5]), fmaxf(rm[6], rm[7])));
  float s = 0.f;
  for (int c = tid; c < L; c += 256) { float e = __expf(pr[c] - m); pr[c] = e; s += e; }
  for (int o = 16; o; o >>= 1) s += __shfl_down(s, o, 32);
  if ((tid & 31) == 0) rsum[tid >> 5] = s;
  __syncthreads();
  float inv = 1.f / (rsum[0]+rsum[1]+rsum[2]+rsum[3]+rsum[4]+rsum[5]+rsum[6]+rsum[7]);
  for (int c = tid; c < L; c += 256) pr[c] *= inv;
}

// Landmark means: ql/kl[b,h,m,d] = mean over LGRP rows of q/k (raw, unscaled).
__global__ __launch_bounds__(256) void landmarks_k(const float* __restrict__ qkv,
                                                   float* __restrict__ ql,
                                                   float* __restrict__ kl) {
  long long idx = (long long)blockIdx.x * 256 + threadIdx.x;   // b,h,m,d
  int d = idx & 63; long long t = idx >> 6;
  int m = t & 255; t >>= 8;
  int hh = t & 7;  int bb = (int)(t >> 3);
  const float* base = qkv + (long long)bb * NPD * QKVD + hh * DH + d;
  long long r0 = (long long)m * LGRP;
  float sq = 0.f, sk = 0.f;
  for (int j = 0; j < LGRP; ++j) {
    const float* pr = base + (r0 + j) * QKVD;
    sq += pr[0]; sk += pr[DIM];
  }
  ql[idx] = sq * (1.f / LGRP);
  kl[idx] = sk * (1.f / LGRP);
}

// E = c*I - X   (batched 256x256)
__global__ __launch_bounds__(256) void eltw_k(const float* __restrict__ X,
                                              float* __restrict__ E, float cdiag) {
  long long idx = (long long)blockIdx.x * 256 + threadIdx.x;
  int j = idx & 255, i = (int)((idx >> 8) & 255);
  E[idx] = (i == j ? cdiag : 0.f) - X[idx];
}

// pinv scaling: per-matrix max row-abs-sum and max col-abs-sum.
__global__ __launch_bounds__(256) void pinv_prep_k(const float* __restrict__ P,
                                                   float* __restrict__ red) {
  const int z = blockIdx.x, tid = threadIdx.x;
  const float* x = P + (long long)z * 65536;
  float rs = 0.f, cs = 0.f;
  for (int k = 0; k < 256; ++k) {
    rs += fabsf(x[tid * 256 + k]);
    cs += fabsf(x[k * 256 + tid]);
  }
  for (int o = 16; o; o >>= 1) {
    rs = fmaxf(rs, __shfl_down(rs, o, 32));
    cs = fmaxf(cs, __shfl_down(cs, o, 32));
  }
  __shared__ float a[8], b[8];
  if ((tid & 31) == 0) { a[tid >> 5] = rs; b[tid >> 5] = cs; }
  __syncthreads();
  if (tid == 0) {
    float m1 = a[0], m2 = b[0];
    for (int i = 1; i < 8; ++i) { m1 = fmaxf(m1, a[i]); m2 = fmaxf(m2, b[i]); }
    red[z] = m1; red[32 + z] = m2;
  }
}

__global__ void pinv_scalar_k(float* red) {
  if (threadIdx.x == 0) {
    float m1 = 0.f, m2 = 0.f;
    for (int i = 0; i < 32; ++i) { m1 = fmaxf(m1, red[i]); m2 = fmaxf(m2, red[32 + i]); }
    red[64] = 1.f / (m1 * m2);
  }
}

// Z = X^T / s   (batched 256x256 transpose scaled)
__global__ __launch_bounds__(256) void pinv_init_k(const float* __restrict__ P,
                                                   float* __restrict__ Z,
                                                   const float* __restrict__ red) {
  float invs = red[64];
  long long idx = (long long)blockIdx.x * 256 + threadIdx.x;
  long long z = idx >> 16;
  int i = (int)((idx >> 8) & 255), j = (int)(idx & 255);
  Z[idx] = P[(z << 16) + ((long long)j << 8) + i] * invs;
}

// Depthwise conv residual over sequence: attnout[b,n,h*64+d] += sum_k v*w[h,k]
__global__ __launch_bounds__(256) void conv_add_k(const float* __restrict__ qkv,
                                                  const float* __restrict__ w,
                                                  float* __restrict__ attnout) {
  long long idx = (long long)blockIdx.x * 256 + threadIdx.x;  // b, n, c
  int c = idx & 511; long long t = idx >> 9;
  int n  = (int)(t % NPD);
  int bb = (int)(t / NPD);
  int hh = c >> 6, d = c & 63;
  const float* vb = qkv + (long long)bb * NPD * QKVD + 2 * DIM + hh * DH + d;
  float acc = 0.f;
#pragma unroll
  for (int k = 0; k < KSZ; ++k) {
    int nn = n + k - (KSZ / 2);
    if (nn >= 0 && nn < NPD) acc += vb[(long long)nn * QKVD] * w[hh * KSZ + k];
  }
  attnout[idx] += acc;
}

// h[b,0,:] = cls_token
__global__ __launch_bounds__(256) void write_cls_k(const float* __restrict__ cls,
                                                   float* __restrict__ h) {
  int e = blockIdx.x * 256 + threadIdx.x;
  int bb = e >> 9, d = e & 511;
  h[(long long)bb * NP * DIM + d] = cls[d];
}

// logits[b,c] = dot(clsln[b], fc2_w[c]) + fc2_b[c]; 8 waves -> 8 outputs
__global__ __launch_bounds__(256) void logits_k(const float* __restrict__ clsln,
                                                const float* __restrict__ w,
                                                const float* __restrict__ bias,
                                                float* __restrict__ out) {
  int tid = threadIdx.x, wv = tid >> 5, lane = tid & 31;
  int bb = wv >> 1, c = wv & 1;
  const float* x = clsln + bb * DIM;
  const float* ww = w + c * DIM;
  float s = 0.f;
  for (int k = lane; k < DIM; k += 32) s += x[k] * ww[k];
  for (int o = 16; o; o >>= 1) s += __shfl_down(s, o, 32);
  if (lane == 0) out[bb * 2 + c] = s + bias[c];
}

// ---------------------------------------------------------------------------
// Host-side orchestration
// ---------------------------------------------------------------------------
struct Ws {
  float *h, *xpad, *qkv, *ql, *kl, *sim1, *sim3, *sim2;
  float *zA, *zB, *xz, *ebuf, *tmp, *tbuf, *t2buf, *attnout, *red, *clsln;
};

static void attn_layer(hipStream_t st, const Ws& w,
                       const float* ng, const float* nb,
                       const float* qkv_w, const float* out_w,
                       const float* out_b, const float* conv_w) {
  const long long sQb = (long long)NPD * QKVD;
  // x = pad(LN(h))
  ln_pad_k<<<dim3(NPD, BDIM), 256, 0, st>>>(w.h, ng, nb, w.xpad);
  // qkv = x @ qkv_w^T
  launch_gemm<1, 0>(st, w.xpad, qkv_w, w.qkv, nullptr,
                    NPD, QKVD, DIM, DIM, DIM, QKVD,
                    (long long)NPD * DIM, 0, 0, 0, sQb, 0, 1, BDIM, 1.f, 0);
  landmarks_k<<<(BDIM * HEADS * MLM * DH) / 256, 256, 0, st>>>(w.qkv, w.ql, w.kl);
  // sim1 = (q * dh^-.5) @ kl^T  -> alpha folds the q scaling
  launch_gemm<1, 0>(st, w.qkv, w.kl, w.sim1, nullptr,
                    NPD, MLM, DH, QKVD, DH, MLM,
                    sQb, DH, (long long)HEADS * MLM * DH, (long long)MLM * DH,
                    (long long)HEADS * NPD * MLM, (long long)NPD * MLM,
                    HEADS, BDIM * HEADS, 0.125f, 0);
  // sim2 = ql @ kl^T
  launch_gemm<1, 0>(st, w.ql, w.kl, w.sim2, nullptr,
                    MLM, MLM, DH, DH, DH, MLM,
                    (long long)HEADS * MLM * DH, (long long)MLM * DH,
                    (long long)HEADS * MLM * DH, (long long)MLM * DH,
                    (long long)HEADS * MLM * MLM, (long long)MLM * MLM,
                    HEADS, BDIM * HEADS, 0.125f, 0);
  // sim3 = ql @ k^T
  launch_gemm<1, 0>(st, w.ql, w.qkv + DIM, w.sim3, nullptr,
                    MLM, NPD, DH, DH, QKVD, NPD,
                    (long long)HEADS * MLM * DH, (long long)MLM * DH,
                    sQb, DH,
                    (long long)HEADS * MLM * NPD, (long long)MLM * NPD,
                    HEADS, BDIM * HEADS, 0.125f, 0);
  softmax_k<<<BDIM * HEADS * NPD, 256, 0, st>>>(w.sim1, MLM);
  softmax_k<<<BDIM * HEADS * MLM, 256, 0, st>>>(w.sim2, MLM);
  softmax_k<<<BDIM * HEADS * MLM, 256, 0, st>>>(w.sim3, NPD);
  // pinv(a2): z0 = a2^T / (max rowsum * max colsum), 6 Newton-Schulz-style iters
  pinv_prep_k<<<32, 256, 0, st>>>(w.sim2, w.red);
  pinv_scalar_k<<<1, 32, 0, st>>>(w.red);
  pinv_init_k<<<(32 * 65536) / 256, 256, 0, st>>>(w.sim2, w.zA, w.red);
  float* za = w.zA; float* zb = w.zB;
  const long long sq = 65536;
  for (int it = 0; it < 6; ++it) {
    launch_gemm<0, 0>(st, w.sim2, za, w.xz, nullptr, MLM, MLM, MLM, MLM, MLM, MLM,
                      sq, 0, sq, 0, sq, 0, 1, 32, 1.f, 0);
    eltw_k<<<8192, 256, 0, st>>>(w.xz, w.ebuf, 7.f);
    launch_gemm<0, 0>(st, w.xz, w.ebuf, w.tmp, nullptr, MLM, MLM, MLM, MLM, MLM, MLM,
                      sq, 0, sq, 0, sq, 0, 1, 32, 1.f, 0);
    eltw_k<<<8192, 256, 0, st>>>(w.tmp, w.ebuf, 15.f);
    launch_gemm<0, 0>(st, w.xz, w.ebuf, w.tmp, nullptr, MLM, MLM, MLM, MLM, MLM, MLM,
                      sq, 0, sq, 0, sq, 0, 1, 32, 1.f, 0);
    eltw_k<<<8192, 256, 0, st>>>(w.tmp, w.ebuf, 13.f);
    launch_gemm<0, 0>(st, za, w.ebuf, zb, nullptr, MLM, MLM, MLM, MLM, MLM, MLM,
                      sq, 0, sq, 0, sq, 0, 1, 32, 0.25f, 0);
    float* tp = za; za = zb; zb = tp;
  }
  // t = a3 @ v
  launch_gemm<0, 0>(st, w.sim3, w.qkv + 2 * DIM, w.tbuf, nullptr,
                    MLM, DH, NPD, NPD, QKVD, DH,
                    (long long)HEADS * MLM * NPD, (long long)MLM * NPD,
                    sQb, DH,
                    (long long)HEADS * MLM * DH, (long long)MLM * DH,
                    HEADS, BDIM * HEADS, 1.f, 0);
  // t2 = a2inv @ t
  launch_gemm<0, 0>(st, za, w.tbuf, w.t2buf, nullptr,
                    MLM, DH, MLM, MLM, DH, DH,
                    sq, 0, (long long)MLM * DH, 0, (long long)MLM * DH, 0,
                    1, 32, 1.f, 0);
  // out = a1 @ t2  (written head-merged into attnout[b, n, h*64+d])
  launch_gemm<0, 0>(st, w.sim1, w.t2buf, w.attnout, nullptr,
                    NPD, DH, MLM, MLM, DH, DIM,
                    (long long)HEADS * NPD * MLM, (long long)NPD * MLM,
                    (long long)HEADS * MLM * DH, (long long)MLM * DH,
                    (long long)NPD * DIM, DH,
                    HEADS, BDIM * HEADS, 1.f, 0);
  // depthwise conv residual on v
  conv_add_k<<<((long long)BDIM * NPD * DIM) / 256, 256, 0, st>>>(w.qkv, conv_w, w.attnout);
  // h += (attnout @ out_w^T + out_b), dropping the PADL left-pad rows
  launch_gemm<1, 2>(st, w.attnout, out_w, w.h, out_b,
                    NPD, DIM, DIM, DIM, DIM, DIM,
                    (long long)NPD * DIM, 0, 0, 0, (long long)NP * DIM, 0,
                    1, BDIM, 1.f, PADL);
}

extern "C" void kernel_launch(void* const* d_in, const int* in_sizes, int n_in,
                              void* d_out, int out_size, void* d_ws, size_t ws_size,
                              hipStream_t stream) {
  const float* data   = (const float*)d_in[0];
  const float* cls    = (const float*)d_in[1];
  const float* fc1_w  = (const float*)d_in[2];
  const float* fc1_b  = (const float*)d_in[3];
  const float* l1ng   = (const float*)d_in[4];
  const float* l1nb   = (const float*)d_in[5];
  const float* l1qkv  = (const float*)d_in[6];
  const float* l1ow   = (const float*)d_in[7];
  const float* l1ob   = (const float*)d_in[8];
  const float* l1cw   = (const float*)d_in[9];
  const float* l2ng   = (const float*)d_in[10];
  const float* l2nb   = (const float*)d_in[11];
  const float* l2qkv  = (const float*)d_in[12];
  const float* l2ow   = (const float*)d_in[13];
  const float* l2ob   = (const float*)d_in[14];
  const float* l2cw   = (const float*)d_in[15];
  const float* normg  = (const float*)d_in[16];
  const float* normb  = (const float*)d_in[17];
  const float* fc2w   = (const float*)d_in[18];
  const float* fc2b   = (const float*)d_in[19];
  (void)in_sizes; (void)n_in; (void)out_size; (void)ws_size;

  float* W = (float*)d_ws;
  Ws w;
  w.h       = W;                     W += (long long)BDIM * NP * DIM;
  w.xpad    = W;                     W += (long long)BDIM * NPD * DIM;
  w.qkv     = W;                     W += (long long)BDIM * NPD * QKVD;
  w.ql      = W;                     W += (long long)BDIM * HEADS * MLM * DH;
  w.kl      = W;                     W += (long long)BDIM * HEADS * MLM * DH;
  w.sim1    = W;                     W += (long long)BDIM * HEADS * NPD * MLM;
  w.sim3    = W;                     W += (long long)BDIM * HEADS * MLM * NPD;
  w.sim2    = W;                     W += (long long)BDIM * HEADS * MLM * MLM;
  w.zA      = W;                     W += (long long)BDIM * HEADS * MLM * MLM;
  w.zB      = W;                     W += (long long)BDIM * HEADS * MLM * MLM;
  w.xz      = W;                     W += (long long)BDIM * HEADS * MLM * MLM;
  w.ebuf    = W;                     W += (long long)BDIM * HEADS * MLM * MLM;
  w.tmp     = W;                     W += (long long)BDIM * HEADS * MLM * MLM;
  w.tbuf    = W;                     W += (long long)BDIM * HEADS * MLM * DH;
  w.t2buf   = W;                     W += (long long)BDIM * HEADS * MLM * DH;
  w.attnout = W;                     W += (long long)BDIM * NPD * DIM;
  w.red     = W;                     W += 128;
  w.clsln   = W;                     W += BDIM * DIM;

  // h = [cls ; relu(data @ fc1_w^T + fc1_b)]
  write_cls_k<<<(BDIM * DIM) / 256, 256, 0, stream>>>(cls, w.h);
  launch_gemm<1, 1>(stream, data, fc1_w, w.h + DIM, fc1_b,
                    SEQN, DIM, OUTER, OUTER, OUTER, DIM,
                    (long long)SEQN * OUTER, 0, 0, 0, (long long)NP * DIM, 0,
                    1, BDIM, 1.f, 0);

  attn_layer(stream, w, l1ng, l1nb, l1qkv, l1ow, l1ob, l1cw);
  attn_layer(stream, w, l2ng, l2nb, l2qkv, l2ow, l2ob, l2cw);

  ln_row0_k<<<BDIM, 256, 0, stream>>>(w.h, normg, normb, w.clsln);
  logits_k<<<1, 256, 0, stream>>>(w.clsln, fc2w, fc2b, (float*)d_out);
}
